// SemlaSelfAttention_73607149519583
// MI455X (gfx1250) — matrix operand order repaired and checked by reading, not compiled, not run on hardware
//
#include <hip/hip_runtime.h>
#include <hip/hip_bf16.h>

#define NN 256
#define FSTRIDE 264   // halves per LDS row (256 + 8 pad -> conflict-free b128 reads)
#define HSTRIDE 264

typedef __attribute__((ext_vector_type(16))) __bf16          v16bf;
typedef __attribute__((ext_vector_type(8)))  float           v8f;
typedef __attribute__((ext_vector_type(16))) unsigned short  v16u;
typedef __attribute__((ext_vector_type(8)))  unsigned short  v8u;

static __device__ __forceinline__ unsigned short f2bf(float f) {
  unsigned int u = __float_as_uint(f);
  u += 0x7FFFu + ((u >> 16) & 1u);
  return (unsigned short)(u >> 16);
}

static __device__ __forceinline__ v8f wmma_bf16(v16bf a, v16bf b, v8f c) {
  // (neg_a, A, neg_b, B, c_mod, C, reuse_a, reuse_b)
  return __builtin_amdgcn_wmma_f32_16x16x32_bf16(false, a, false, b, (short)0, c, false, false);
}

// A fragment (16x32 bf16) from LDS, row-major rows with given stride.
// Lane l: row = base + (l%16); halves 0..7 -> K = kt*32 + (l/16)*8 .. +7,
// halves 8..15 -> +16.  Two 16B ds loads.
static __device__ __forceinline__ v16bf ldA(const unsigned short* s, int stride,
                                            int rowbase, int kt, int lane) {
  int row  = rowbase + (lane & 15);
  int koff = (kt << 5) + ((lane >> 4) << 3);
  const unsigned short* p = s + row * stride + koff;
  v8u lo = *(const v8u*)(p);
  v8u hi = *(const v8u*)(p + 16);
  v16u r;
#pragma unroll
  for (int i = 0; i < 8; ++i) { r[i] = lo[i]; r[i + 8] = hi[i]; }
  return __builtin_bit_cast(v16bf, r);
}

// B fragment (32x16 bf16) from pre-swizzled weights: per-lane 16 contiguous halves.
static __device__ __forceinline__ v16bf ldB(const unsigned short* w, int nt, int kt, int lane) {
  const v16u* p = (const v16u*)(w + ((((nt * 8) + kt) * 32) + lane) * 16);
  return __builtin_bit_cast(v16bf, *p);
}

// ---------------- prep: swizzle mW1/mW2 into WMMA B-fragment order ----------------
extern "C" __global__ void semla_prep_w(const float* mW1, const float* mW2,
                                        unsigned short* wB1, unsigned short* wB2) {
  int i = blockIdx.x * blockDim.x + threadIdx.x;
  if (i < 65536) {  // mW1: 16 nt * 8 kt * 32 lanes * 16 halves
    int h = i & 15, lane = (i >> 4) & 31, kt = (i >> 9) & 7, nt = i >> 12;
    int k = (kt << 5) + ((h >> 3) << 4) + ((lane >> 4) << 3) + (h & 7);
    int n = (nt << 4) + (lane & 15);
    wB1[i] = f2bf(mW1[k * 256 + n]);
  }
  int j = i - 65536;
  if (j >= 0 && j < 40960) {  // mW2: 10 nt * 8 kt * 32 * 16
    int h = j & 15, lane = (j >> 4) & 31, kt = (j >> 9) & 7, nt = j >> 12;
    int k = (kt << 5) + ((h >> 3) << 4) + ((lane >> 4) << 3) + (h & 7);
    int n = (nt << 4) + (lane & 15);
    wB2[j] = f2bf(mW2[k * 160 + n]);
  }
}

// ---------------- prep: per-node projections qm/km/hf/pe ----------------
extern "C" __global__ __launch_bounds__(256)
void semla_prep_rows(const float* equis, const float* invs,
                     const float* qW, const float* qb, const float* kW, const float* kb,
                     const float* inv_inW, const float* inv_inb, const float* equi_inW,
                     unsigned short* qm_bf, unsigned short* km_bf, float* hf, float* pe) {
  __shared__ float sInv[256];
  const int row = blockIdx.x;          // b*256 + n
  const int t   = threadIdx.x;
  sInv[t] = invs[(size_t)row * 256 + t];
  __syncthreads();
  {
    float acc = inv_inb[t];
    for (int k = 0; k < 256; ++k) acc += sInv[k] * inv_inW[k * 256 + t];
    hf[(size_t)row * 256 + t] = acc;
  }
  if (t < 64) {
    float aq = qb[t], ak = kb[t];
    for (int k = 0; k < 256; ++k) { aq += sInv[k] * qW[k * 64 + t]; ak += sInv[k] * kW[k * 64 + t]; }
    qm_bf[row * 64 + t] = f2bf(aq);
    km_bf[row * 64 + t] = f2bf(ak);
  }
  if (t < 192) {
    int d = t >> 6, c = t & 63;
    const float* eq = equis + ((size_t)row * 3 + d) * 64;
    float acc = 0.f;
    for (int j = 0; j < 64; ++j) acc += eq[j] * equi_inW[j * 64 + c];
    pe[(size_t)row * 192 + t] = acc;
  }
}

// ---------------- main fused kernel: one block per (b,q) ----------------
extern "C" __global__ __launch_bounds__(256)
void semla_main(const float* __restrict__ equis, const float* __restrict__ edges,
                const int* __restrict__ adj,
                const unsigned short* __restrict__ qm_bf, const unsigned short* __restrict__ km_bf,
                const float* __restrict__ hf, const float* __restrict__ pe,
                const unsigned short* __restrict__ wB1, const unsigned short* __restrict__ wB2,
                const float* __restrict__ mb1, const float* __restrict__ mb2,
                const float* __restrict__ inv_outW, const float* __restrict__ inv_outb,
                const float* __restrict__ equi_outW,
                float* __restrict__ out_equi, float* __restrict__ out_inv,
                float* __restrict__ out_edge) {
  extern __shared__ char smem[];
  unsigned short* sFeats = (unsigned short*)smem;                       // 256*264*2 = 135168 B
  unsigned short* sHid   = (unsigned short*)(smem + 256 * FSTRIDE * 2); // 135168 B
  float* sMsg  = (float*)smem;                                          // 256*96*4 = 98304 B (overlays sFeats)
  float* sAux  = (float*)(smem + 2 * 256 * FSTRIDE * 2);
  float* sEqQ  = sAux;         // 192
  float* sMask = sAux + 192;   // 256
  float* sW    = sAux + 448;   // 96
  float* sVec  = sAux + 544;   // 256

  const int t    = threadIdx.x;
  const int lane = t & 31;
  const int wave = t >> 5;
  const int bq   = blockIdx.x;
  const int b    = bq >> 8;
  const int mbase = wave << 5;     // 32-row M strip per wave

  __builtin_prefetch(wB1 + (t << 8), 0, 3);
  __builtin_prefetch(wB2 + t * 160, 0, 3);

  // equis[b,q,:,:] -> LDS
  if (t < 192) sEqQ[t] = equis[(size_t)bq * 192 + t];
  __syncthreads();

  // ---- build feats tile (256 k-rows x 256 feature cols, bf16) ----
  {
    unsigned short qv = (t < 64) ? qm_bf[bq * 64 + t] : 0;
    const int c128 = t - 128;
    for (int r = 0; r < NN; ++r) {
      unsigned short v;
      if (t < 64) {
        v = qv;
      } else if (t < 128) {
        v = km_bf[(((b << 8) + r) << 6) + (t - 64)];
      } else if (t < 192) {
        const float* eqk = equis + (size_t)((b << 8) + r) * 192;
        float d0 = sEqQ[c128] * eqk[c128] + sEqQ[64 + c128] * eqk[64 + c128]
                 + sEqQ[128 + c128] * eqk[128 + c128];
        v = f2bf(d0);
      } else {
        v = f2bf(edges[(((size_t)bq << 8) + r) * 64 + (t - 192)]);
      }
      sFeats[r * FSTRIDE + t] = v;
    }
  }
  __syncthreads();

  // ---- GEMM1: feats(32x256 strip) @ mW1(256x256) -> SiLU -> sHid (bf16) ----
  for (int ng = 0; ng < 4; ++ng) {
    v8f acc[2][4];
#pragma unroll
    for (int i = 0; i < 2; ++i)
#pragma unroll
      for (int j = 0; j < 4; ++j) acc[i][j] = (v8f)(0.f);
    for (int kt = 0; kt < 8; ++kt) {
      v16bf a0 = ldA(sFeats, FSTRIDE, mbase,      kt, lane);
      v16bf a1 = ldA(sFeats, FSTRIDE, mbase + 16, kt, lane);
#pragma unroll
      for (int j = 0; j < 4; ++j) {
        v16bf bb = ldB(wB1, (ng << 2) + j, kt, lane);
        acc[0][j] = wmma_bf16(a0, bb, acc[0][j]);
        acc[1][j] = wmma_bf16(a1, bb, acc[1][j]);
      }
    }
#pragma unroll
    for (int i = 0; i < 2; ++i)
#pragma unroll
      for (int j = 0; j < 4; ++j) {
        int col  = (((ng << 2) + j) << 4) | (lane & 15);
        float bias = mb1[col];
        int rbase = mbase + (i << 4) + ((lane >> 4) << 3);
#pragma unroll
        for (int g = 0; g < 8; ++g) {
          float x = acc[i][j][g] + bias;
          float h = x / (1.f + __expf(-x));          // SiLU
          sHid[(rbase + g) * HSTRIDE + col] = f2bf(h);
        }
      }
  }
  __syncthreads();   // all waves done with sFeats -> safe to overlay sMsg

  // ---- GEMM2: hidden(32x256) @ mW2(256x160); cols<96 -> sMsg, cols>=96 -> edge out ----
  float* edge_base = out_edge + ((size_t)bq << 8) * 64;
  for (int pass = 0; pass < 2; ++pass) {
    v8f acc[2][5];
#pragma unroll
    for (int i = 0; i < 2; ++i)
#pragma unroll
      for (int j = 0; j < 5; ++j) acc[i][j] = (v8f)(0.f);
    for (int kt = 0; kt < 8; ++kt) {
      v16bf a0 = ldA(sHid, HSTRIDE, mbase,      kt, lane);
      v16bf a1 = ldA(sHid, HSTRIDE, mbase + 16, kt, lane);
#pragma unroll
      for (int j = 0; j < 5; ++j) {
        v16bf bb = ldB(wB2, pass * 5 + j, kt, lane);
        acc[0][j] = wmma_bf16(a0, bb, acc[0][j]);
        acc[1][j] = wmma_bf16(a1, bb, acc[1][j]);
      }
    }
#pragma unroll
    for (int i = 0; i < 2; ++i)
#pragma unroll
      for (int j = 0; j < 5; ++j) {
        int col  = (pass * 5 + j) * 16 + (lane & 15);
        float bias = mb2[col];
        int rbase = mbase + (i << 4) + ((lane >> 4) << 3);
#pragma unroll
        for (int g = 0; g < 8; ++g) {
          float val = acc[i][j][g] + bias;
          int r = rbase + g;
          if (col < 96) sMsg[r * 96 + col] = val;
          else          edge_base[(size_t)r * 64 + (col - 96)] = val;
        }
      }
  }

  // ---- attention over k (rows) ----
  sMask[t] = (adj[((size_t)bq << 8) + t] == 0) ? -1e30f : 0.f;
  __syncthreads();

  if (t < 96) {
    const int c = t;
    float mx = -1e30f;
    for (int k = 0; k < NN; ++k) mx = fmaxf(mx, sMsg[k * 96 + c] + sMask[k]);
    float sum = 0.f;
    for (int k = 0; k < NN; ++k) {
      float e = __expf(sMsg[k * 96 + c] + sMask[k] - mx);
      sMsg[k * 96 + c] = e;
      sum += e;
    }
    float inv = 1.f / sum, s2 = 0.f;
    for (int k = 0; k < NN; ++k) {
      float a = sMsg[k * 96 + c] * inv;
      sMsg[k * 96 + c] = a;
      s2 += a * a;
    }
    sW[c] = sqrtf(s2);
  }
  __syncthreads();

  // out_i = einsum(attn_i, hf) * w_i   (thread t = h*8+d, coalesced hf reads)
  {
    const int h = t >> 3;
    float acc = 0.f;
    for (int k = 0; k < NN; ++k)
      acc += sMsg[k * 96 + h] * hf[((((size_t)b << 8) + k) << 8) + t];
    sVec[t] = acc * sW[h];
  }
  __syncthreads();
  // inv_updates = out_i @ inv_outW + b
  {
    float acc = inv_outb[t];
    for (int r = 0; r < 256; ++r) acc += sVec[r] * inv_outW[r * 256 + t];
    out_inv[((size_t)bq << 8) + t] = acc;
  }
  __syncthreads();
  // out_e = einsum(attn_e, pe) * w_e   (thread t = d*64+c)
  if (t < 192) {
    const int c = t & 63;
    float acc = 0.f;
    for (int k = 0; k < NN; ++k)
      acc += sMsg[k * 96 + 32 + c] * pe[(((size_t)b << 8) + k) * 192 + t];
    sVec[t] = acc * sW[32 + c];
  }
  __syncthreads();
  // equi_updates = out_e @ equi_outW
  if (t < 192) {
    const int d = t >> 6, o = t & 63;
    const float* oe = sVec + (d << 6);
    float acc = 0.f;
    for (int c = 0; c < 64; ++c) acc += oe[c] * equi_outW[c * 64 + o];
    out_equi[(size_t)bq * 192 + (d << 6) + o] = acc;
  }
}

extern "C" void kernel_launch(void* const* d_in, const int* in_sizes, int n_in,
                              void* d_out, int out_size, void* d_ws, size_t ws_size,
                              hipStream_t stream) {
  const float* equis     = (const float*)d_in[0];
  const float* invs      = (const float*)d_in[1];
  const float* edges     = (const float*)d_in[2];
  const int*   adj       = (const int*)d_in[3];
  const float* qW        = (const float*)d_in[4];
  const float* qb        = (const float*)d_in[5];
  const float* kW        = (const float*)d_in[6];
  const float* kb        = (const float*)d_in[7];
  const float* mW1       = (const float*)d_in[8];
  const float* mb1       = (const float*)d_in[9];
  const float* mW2       = (const float*)d_in[10];
  const float* mb2       = (const float*)d_in[11];
  const float* inv_inW   = (const float*)d_in[12];
  const float* inv_inb   = (const float*)d_in[13];
  const float* inv_outW  = (const float*)d_in[14];
  const float* inv_outb  = (const float*)d_in[15];
  const float* equi_inW  = (const float*)d_in[16];
  const float* equi_outW = (const float*)d_in[17];

  float* out_equi = (float*)d_out;                    // B*N*3*64 = 196608
  float* out_inv  = out_equi + 196608;                // B*N*256  = 262144
  float* out_edge = out_inv + 262144;                 // B*N*N*64 = 16777216

  char* ws = (char*)d_ws;
  unsigned short* qm_bf = (unsigned short*)(ws);            // 131072 B
  unsigned short* km_bf = (unsigned short*)(ws + 131072);   // 131072 B
  unsigned short* wB1   = (unsigned short*)(ws + 262144);   // 131072 B
  unsigned short* wB2   = (unsigned short*)(ws + 393216);   //  81920 B
  float*          hf    = (float*)(ws + 475136);            // 1048576 B
  float*          pe    = (float*)(ws + 1523712);           //  786432 B  (end: 2310144)

  semla_prep_w<<<(65536 + 40960 + 255) / 256, 256, 0, stream>>>(mW1, mW2, wB1, wB2);
  semla_prep_rows<<<1024, 256, 0, stream>>>(equis, invs, qW, qb, kW, kb,
                                            inv_inW, inv_inb, equi_inW,
                                            qm_bf, km_bf, hf, pe);

  size_t smem = (size_t)2 * 256 * FSTRIDE * 2 + 800 * 4;  // 273536 B < 320 KB/WGP
  semla_main<<<1024, 256, smem, stream>>>(equis, edges, adj, qm_bf, km_bf, hf, pe,
                                          wB1, wB2, mb1, mb2,
                                          inv_outW, inv_outb, equi_outW,
                                          out_equi, out_inv, out_edge);
}
